// Y_Compression_Model_25520695673046
// MI455X (gfx1250) — matrix operand order repaired and checked by reading, not compiled, not run on hardware
//
#include <hip/hip_runtime.h>
#include <hip/hip_bf16.h>

// ---------------- problem constants ----------------
#define VOCAB   50257
#define EMB_DIM 768
#define NUM_SEQ 3
#define SEQ_LEN 312
#define BATCH   2048
#define HDIM    (EMB_DIM * NUM_SEQ)   // 2304

// ---------------- CDNA5 vector types ----------------
typedef __attribute__((ext_vector_type(16))) __bf16 v16bf;
typedef __attribute__((ext_vector_type(8)))  __bf16 v8bf;
typedef __attribute__((ext_vector_type(8)))  float  v8f;

union FragBF { v16bf v; v8bf h[2]; };

__device__ __forceinline__ unsigned short f2bf(float f) {
  // round-to-nearest-even f32 -> bf16
  unsigned int u = __float_as_uint(f);
  u += 0x7FFFu + ((u >> 16) & 1u);
  return (unsigned short)(u >> 16);
}

// ---------------- f32 -> bf16 weight conversion ----------------
__global__ void f32_to_bf16_kernel(const float* __restrict__ in,
                                   unsigned short* __restrict__ out, int n) {
  int i = blockIdx.x * blockDim.x + threadIdx.x;
  if (i < n) out[i] = f2bf(in[i]);
}

// ---------------- embedding gather + mean ----------------
// One block per (batch, seq). 192 threads, each owns 4 consecutive dims
// (float4 -> global_load_b128, fully coalesced). Token ids staged in LDS.
__global__ void gather_mean_kernel(const int* __restrict__ x,
                                   const float* __restrict__ emb,
                                   unsigned short* __restrict__ A) {
  __shared__ int toks[SEQ_LEN];
  const int b = blockIdx.x / NUM_SEQ;
  const int s = blockIdx.x % NUM_SEQ;

  const int* xrow = x + (size_t)b * (NUM_SEQ * SEQ_LEN) + s * SEQ_LEN;
  for (int t = threadIdx.x; t < SEQ_LEN; t += blockDim.x) {
    int id = xrow[t];
    id = id < 0 ? 0 : (id >= VOCAB ? VOCAB - 1 : id);
    toks[t] = id;
  }
  __syncthreads();

  const int d4 = threadIdx.x * 4;   // 0..764, covers 768 dims
  float4 acc = make_float4(0.f, 0.f, 0.f, 0.f);
  for (int t = 0; t < SEQ_LEN; ++t) {
    const float4 e = *(const float4*)(emb + (size_t)toks[t] * EMB_DIM + d4);
    acc.x += e.x; acc.y += e.y; acc.z += e.z; acc.w += e.w;
  }
  const float inv = 1.0f / (float)SEQ_LEN;
  unsigned short* arow = A + (size_t)b * HDIM + s * EMB_DIM + d4;
  unsigned int p0 = (unsigned int)f2bf(acc.x * inv) | ((unsigned int)f2bf(acc.y * inv) << 16);
  unsigned int p1 = (unsigned int)f2bf(acc.z * inv) | ((unsigned int)f2bf(acc.w * inv) << 16);
  *(uint2*)arow = make_uint2(p0, p1);
}

// ---------------- WMMA GEMM: out = act(A[M,K] * W[N,K]^T + bias) ----------------
// One wave computes one 16x16 output tile. bf16 inputs, f32 accumulate via
// v_wmma_f32_16x16x32_bf16. Dual accumulators (K step 64) break the WMMA RAW chain.
template<int M, int N, int K, bool RELU, bool OUT_BF16>
__global__ void wmma_gemm_kernel(const __bf16* __restrict__ A,
                                 const __bf16* __restrict__ W,
                                 const float* __restrict__ bias,
                                 void* __restrict__ outp) {
  constexpr int TN = N / 16;
  const int wave  = (int)((blockIdx.x * blockDim.x + threadIdx.x) >> 5);
  const int lane  = (int)(threadIdx.x & 31);
  const int mtile = wave / TN;
  const int ntile = wave % TN;
  if (mtile >= M / 16) return;           // wave-uniform: EXEC stays all-ones

  const int half = lane >> 4;            // which 16-lane half
  const int r    = lane & 15;

  const __bf16* arow = A + (size_t)(mtile * 16 + r) * K;
  const __bf16* wrow = W + (size_t)(ntile * 16 + r) * K;

  v8f acc0 = {0.f, 0.f, 0.f, 0.f, 0.f, 0.f, 0.f, 0.f};
  v8f acc1 = {0.f, 0.f, 0.f, 0.f, 0.f, 0.f, 0.f, 0.f};

  for (int kb = 0; kb < K; kb += 64) {
    // ---- block 0: K = kb .. kb+31 ----
    {
      FragBF fa, fb;
      // A 16x32 bf16 layout: half0 -> K {0..7, 16..23}, half1 -> {8..15, 24..31}
      fa.h[0] = *(const v8bf*)(arow + kb + half * 8);
      fa.h[1] = *(const v8bf*)(arow + kb + 16 + half * 8);
      // B 32x16 bf16 layout: half0 -> K {0..7, 16..23}, half1 -> {8..15, 24..31}
      fb.h[0] = *(const v8bf*)(wrow + kb + half * 8);
      fb.h[1] = *(const v8bf*)(wrow + kb + 16 + half * 8);
      acc0 = __builtin_amdgcn_wmma_f32_16x16x32_bf16(
          false, fa.v, false, fb.v, (short)0, acc0, false, false);
    }
    // ---- block 1: K = kb+32 .. kb+63 ----
    {
      const int kc = kb + 32;
      FragBF fa, fb;
      fa.h[0] = *(const v8bf*)(arow + kc + half * 8);
      fa.h[1] = *(const v8bf*)(arow + kc + 16 + half * 8);
      fb.h[0] = *(const v8bf*)(wrow + kc + half * 8);
      fb.h[1] = *(const v8bf*)(wrow + kc + 16 + half * 8);
      acc1 = __builtin_amdgcn_wmma_f32_16x16x32_bf16(
          false, fa.v, false, fb.v, (short)0, acc1, false, false);
    }
  }
  v8f acc = acc0 + acc1;

  // C/D layout: VGPR vr -> row (vr + half*8), col (lane & 15)
  const int   col = ntile * 16 + r;
  const float bv  = bias[col];
#pragma unroll
  for (int vr = 0; vr < 8; ++vr) {
    const int row = mtile * 16 + vr + half * 8;
    float v = acc[vr] + bv;
    if (RELU) v = v > 0.f ? v : 0.f;
    if (OUT_BF16) {
      ((unsigned short*)outp)[(size_t)row * N + col] = f2bf(v);
    } else {
      ((float*)outp)[(size_t)row * N + col] = v;
    }
  }
}

// ---------------- launch ----------------
extern "C" void kernel_launch(void* const* d_in, const int* in_sizes, int n_in,
                              void* d_out, int out_size, void* d_ws, size_t ws_size,
                              hipStream_t stream) {
  const int*   x   = (const int*)d_in[0];
  const float* emb = (const float*)d_in[1];
  const float* w1  = (const float*)d_in[2];
  const float* b1  = (const float*)d_in[3];
  const float* w2  = (const float*)d_in[4];
  const float* b2  = (const float*)d_in[5];
  const float* w3  = (const float*)d_in[6];
  const float* b3  = (const float*)d_in[7];
  float* out = (float*)d_out;

  // workspace carve-out (256B aligned regions), ~12 MB total
  char* ws = (char*)d_ws;
  size_t off = 0;
  auto carve = [&](size_t bytes) -> void* {
    void* p = ws + off;
    off = (off + bytes + 255) & ~(size_t)255;
    return p;
  };
  unsigned short* Abf  = (unsigned short*)carve((size_t)BATCH * HDIM * 2); // [2048,2304] bf16
  unsigned short* W1bf = (unsigned short*)carve((size_t)256 * HDIM * 2);   // [256,2304]  bf16
  unsigned short* W2bf = (unsigned short*)carve((size_t)64 * 256 * 2);     // [64,256]    bf16
  unsigned short* W3bf = (unsigned short*)carve((size_t)32 * 64 * 2);      // [32,64]     bf16
  unsigned short* H1bf = (unsigned short*)carve((size_t)BATCH * 256 * 2);  // [2048,256]  bf16
  unsigned short* H2bf = (unsigned short*)carve((size_t)BATCH * 64 * 2);   // [2048,64]   bf16

  // 1) weights -> bf16
  f32_to_bf16_kernel<<<(256 * HDIM + 255) / 256, 256, 0, stream>>>(w1, W1bf, 256 * HDIM);
  f32_to_bf16_kernel<<<(64 * 256 + 255) / 256, 256, 0, stream>>>(w2, W2bf, 64 * 256);
  f32_to_bf16_kernel<<<(32 * 64 + 255) / 256, 256, 0, stream>>>(w3, W3bf, 32 * 64);

  // 2) gather + mean (L2-resident table; b128 coalesced gathers)
  gather_mean_kernel<<<BATCH * NUM_SEQ, 192, 0, stream>>>(x, emb, Abf);

  // 3) MLP via WMMA; waves = (M/16)*(N/16), 8 waves per 256-thread block
  wmma_gemm_kernel<BATCH, 256, HDIM, true,  true >
      <<<(BATCH / 16) * (256 / 16) / 8, 256, 0, stream>>>(
          (const __bf16*)Abf, (const __bf16*)W1bf, b1, H1bf);
  wmma_gemm_kernel<BATCH, 64, 256, true,  true >
      <<<(BATCH / 16) * (64 / 16) / 8, 256, 0, stream>>>(
          (const __bf16*)H1bf, (const __bf16*)W2bf, b2, H2bf);
  wmma_gemm_kernel<BATCH, 32, 64, false, false>
      <<<(BATCH / 16) * (32 / 16) / 8, 256, 0, stream>>>(
          (const __bf16*)H2bf, (const __bf16*)W3bf, b3, out);
}
